// RNNDecoder_14843406974995
// MI455X (gfx1250) — matrix operand order repaired and checked by reading, not compiled, not run on hardware
//
#include <hip/hip_runtime.h>
#include <hip/hip_bf16.h>
#include <math.h>

// ---------------------------------------------------------------------------
// RNN autoregressive decoder for MI455X (gfx1250, wave32, WMMA).
//   B=4096, BOTTLENECK=32, HIDDEN=512, OUT_DIM=1024, SEQ_LEN=50
// Core GEMMs: v_wmma_f32_16x16x32_bf16 (bf16 A/B, f32 accumulate).
// 4 waves / block (2x2 wave tiles), each wave owns a 32x64 output tile.
// ---------------------------------------------------------------------------

typedef __bf16 bf16_t;
typedef __attribute__((ext_vector_type(16))) __bf16 v16bf;
typedef __attribute__((ext_vector_type(8)))  float  v8f;
typedef __attribute__((ext_vector_type(4)))  float  v4f;

#define RNN_B       4096
#define RNN_BOT     32
#define RNN_H       512
#define RNN_D       1024
#define RNN_T       50

// Hardware tanh if the gfx1250 builtin is available; libm fallback otherwise.
__device__ __forceinline__ float fast_tanh(float x) {
#if __has_builtin(__builtin_amdgcn_tanhf)
  return __builtin_amdgcn_tanhf(x);
#elif __has_builtin(__builtin_amdgcn_tanh_f32)
  return __builtin_amdgcn_tanh_f32(x);
#else
  return tanhf(x);
#endif
}

// ---------------------------------------------------------------------------
// A-fragment loader: 16-bit A matrix 16x32, contiguous K order (matches B):
//   lanes 0-15  (half=0): elems 0..15 -> K=k0+0..15
//   lanes 16-31 (half=1): elems 0..15 -> K=k0+16..31
// rowp = start of this lane's activation row (f32); convert to bf16 in-reg.
// ---------------------------------------------------------------------------
__device__ __forceinline__ v16bf load_a_frag_f32(const float* __restrict__ rowp,
                                                 int k0, int half) {
  const float* g1 = rowp + k0 + 16 * half;       // K block [base, base+8)
  const float* g2 = rowp + k0 + 16 * half + 8;   // K block [base+8, base+16)
  v4f a0 = *(const v4f*)(g1);
  v4f a1 = *(const v4f*)(g1 + 4);
  v4f a2 = *(const v4f*)(g2);
  v4f a3 = *(const v4f*)(g2 + 4);
  v16bf r;
#pragma unroll
  for (int i = 0; i < 4; ++i) {
    r[i]      = (bf16_t)a0[i];
    r[4 + i]  = (bf16_t)a1[i];
    r[8 + i]  = (bf16_t)a2[i];
    r[12 + i] = (bf16_t)a3[i];
  }
  return r;
}

// Accumulate a 32x64 wave tile over one input GEMM: acc += A[32 x K] * W^T.
// W is bf16 row-major [Ncols][K] (row = output column).
__device__ __forceinline__ void gemm_accum(v8f acc[2][4],
                                           const float* __restrict__ A, int sA,
                                           const bf16_t* __restrict__ W, int K,
                                           int row_base, int col_base,
                                           int ln, int half) {
  const float* a_r0 = A + (long)(row_base + ln) * sA;
  const float* a_r1 = a_r0 + (long)16 * sA;
  const bf16_t* wc[4];
#pragma unroll
  for (int ct = 0; ct < 4; ++ct)
    wc[ct] = W + (long)(col_base + 16 * ct + ln) * K + 16 * half;

  for (int k0 = 0; k0 < K; k0 += 32) {
    // Stage all fragments for this K chunk first, then issue 8 WMMAs.
    v16bf a0 = load_a_frag_f32(a_r0, k0, half);
    v16bf a1 = load_a_frag_f32(a_r1, k0, half);
    v16bf b[4];
#pragma unroll
    for (int ct = 0; ct < 4; ++ct)
      b[ct] = *(const v16bf*)(wc[ct] + k0);
#pragma unroll
    for (int ct = 0; ct < 4; ++ct) {
      acc[0][ct] = __builtin_amdgcn_wmma_f32_16x16x32_bf16(
          false, a0, false, b[ct], (short)0, acc[0][ct], false, false);
      acc[1][ct] = __builtin_amdgcn_wmma_f32_16x16x32_bf16(
          false, a1, false, b[ct], (short)0, acc[1][ct], false, false);
    }
  }
}

// ---------------------------------------------------------------------------
// Step GEMM: OUT = act( A1*W1^T (+ A2*W2^T) + bias1 (+ bias2) )
// Block = 128 threads = 4 waves in a 2x2 arrangement; block tile 64 x 128.
// Each wave computes 32 rows x 64 cols = 2x4 grid of 16x16 WMMA tiles.
// ---------------------------------------------------------------------------
__global__ void __launch_bounds__(128)
rnn_step_gemm(const float* __restrict__ A1, int sA1,
              const bf16_t* __restrict__ W1, int K1,
              const float* __restrict__ A2, int sA2,
              const bf16_t* __restrict__ W2, int K2,
              const float* __restrict__ bias1,
              const float* __restrict__ bias2,
              float* __restrict__ OUT, int sOut,
              int apply_tanh) {
  const int l    = threadIdx.x & 31;  // lane; full wave active (WMMA needs EXEC all-1)
  const int wid  = threadIdx.x >> 5;  // wave 0..3
  const int half = l >> 4;
  const int ln   = l & 15;

  const int row_base = blockIdx.y * 64 + (wid >> 1) * 32;
  const int col_base = blockIdx.x * 128 + (wid & 1) * 64;

  v8f acc[2][4];
#pragma unroll
  for (int rt = 0; rt < 2; ++rt)
#pragma unroll
    for (int ct = 0; ct < 4; ++ct) acc[rt][ct] = (v8f)0.0f;

  gemm_accum(acc, A1, sA1, W1, K1, row_base, col_base, ln, half);
  if (A2 != nullptr)
    gemm_accum(acc, A2, sA2, W2, K2, row_base, col_base, ln, half);

  // Epilogue. C/D layout: lane -> column (l&15); VGPR r -> row r + 8*half.
#pragma unroll
  for (int ct = 0; ct < 4; ++ct) {
    const int col = col_base + 16 * ct + ln;
    float bsum = bias1 ? bias1[col] : 0.0f;
    if (bias2) bsum += bias2[col];
#pragma unroll
    for (int rt = 0; rt < 2; ++rt) {
      const int row0 = row_base + 16 * rt + 8 * half;
#pragma unroll
      for (int r = 0; r < 8; ++r) {
        float v = acc[rt][ct][r] + bsum;
        if (apply_tanh) v = fast_tanh(v);
        OUT[(long)(row0 + r) * sOut + col] = v;
      }
    }
  }
}

// ---------------------------------------------------------------------------
// h0 = tanh(z @ W_hp^T + b_hp)   [B x H], K=32 (tiny; plain VALU)
// ---------------------------------------------------------------------------
__global__ void rnn_h0_kernel(const float* __restrict__ z,
                              const float* __restrict__ W_hp,
                              const float* __restrict__ b_hp,
                              float* __restrict__ hbuf) {
  int idx = blockIdx.x * blockDim.x + threadIdx.x;  // B*H threads
  int b = idx >> 9;            // / 512
  int j = idx & (RNN_H - 1);
  const float* zr = z + (long)b * RNN_BOT;
  const float* wr = W_hp + (long)j * RNN_BOT;
  float s = b_hp[j];
#pragma unroll
  for (int k = 0; k < RNN_BOT; ++k) s += zr[k] * wr[k];
  hbuf[idx] = fast_tanh(s);
}

// x0 broadcast: xbuf[b, d] = start_token[d]
__global__ void rnn_x0_kernel(const float* __restrict__ start_token,
                              float* __restrict__ xbuf) {
  int idx = blockIdx.x * blockDim.x + threadIdx.x;  // B*D threads
  xbuf[idx] = start_token[idx & (RNN_D - 1)];
}

// f32 -> bf16 weight conversion (RNE via cast)
__global__ void rnn_cvt_bf16(const float* __restrict__ src,
                             bf16_t* __restrict__ dst, int n) {
  int idx = blockIdx.x * blockDim.x + threadIdx.x;
  if (idx < n) dst[idx] = (bf16_t)src[idx];
}

// ---------------------------------------------------------------------------
extern "C" void kernel_launch(void* const* d_in, const int* in_sizes, int n_in,
                              void* d_out, int out_size, void* d_ws, size_t ws_size,
                              hipStream_t stream) {
  (void)in_sizes; (void)n_in; (void)out_size; (void)ws_size;

  const float* z       = (const float*)d_in[0];
  const float* W_hp    = (const float*)d_in[1];
  const float* b_hp    = (const float*)d_in[2];
  const float* W_ih    = (const float*)d_in[3];
  const float* b_ih    = (const float*)d_in[4];
  const float* W_hh    = (const float*)d_in[5];
  const float* b_hh    = (const float*)d_in[6];
  const float* W_out   = (const float*)d_in[7];
  const float* b_out   = (const float*)d_in[8];
  const float* start_t = (const float*)d_in[9];

  float* out = (float*)d_out;
  float* recon = out;                               // [B, T, D]
  float* hid   = out + (long)RNN_B * RNN_T * RNN_D; // [B, T, H]

  // workspace layout
  char* ws = (char*)d_ws;
  bf16_t* Wih_b  = (bf16_t*)(ws);                                     // 512*1024 bf16 = 1 MiB
  bf16_t* Whh_b  = (bf16_t*)(ws + (1u << 20));                        // 512*512  bf16 = 512 KiB
  bf16_t* Wout_b = (bf16_t*)(ws + (1u << 20) + (512u << 10));         // 1024*512 bf16 = 1 MiB
  float*  xbuf   = (float*)(ws + (5u << 19));                         // B*D f32 = 16 MiB
  float*  hbuf   = (float*)(ws + (5u << 19) + ((size_t)RNN_B * RNN_D * 4)); // B*H f32 = 8 MiB

  const int nWih  = RNN_H * RNN_D;
  const int nWhh  = RNN_H * RNN_H;
  const int nWout = RNN_D * RNN_H;

  // 1) weight conversion f32 -> bf16
  rnn_cvt_bf16<<<(nWih  + 255) / 256, 256, 0, stream>>>(W_ih,  Wih_b,  nWih);
  rnn_cvt_bf16<<<(nWhh  + 255) / 256, 256, 0, stream>>>(W_hh,  Whh_b,  nWhh);
  rnn_cvt_bf16<<<(nWout + 255) / 256, 256, 0, stream>>>(W_out, Wout_b, nWout);

  // 2) initial state
  rnn_h0_kernel<<<(RNN_B * RNN_H) / 256, 256, 0, stream>>>(z, W_hp, b_hp, hbuf);
  rnn_x0_kernel<<<(RNN_B * RNN_D) / 256, 256, 0, stream>>>(start_t, xbuf);

  // 3) sequential decode: 2 WMMA-GEMM kernels per timestep.
  //    Activations flow through d_out itself (pred_t is x_{t+1}, h_t read back).
  const int sRecon = RNN_T * RNN_D;  // row stride inside [B,T,D]
  const int sHid   = RNN_T * RNN_H;  // row stride inside [B,T,H]

  for (int t = 0; t < RNN_T; ++t) {
    const float* xsrc; int sx;
    const float* hsrc; int sh;
    if (t == 0) {
      xsrc = xbuf; sx = RNN_D;
      hsrc = hbuf; sh = RNN_H;
    } else {
      xsrc = recon + (long)(t - 1) * RNN_D; sx = sRecon;
      hsrc = hid   + (long)(t - 1) * RNN_H; sh = sHid;
    }
    float* h_t = hid   + (long)t * RNN_H;
    float* p_t = recon + (long)t * RNN_D;

    // h_t = tanh(x W_ih^T + h_prev W_hh^T + b_ih + b_hh)   [4096 x 512]
    rnn_step_gemm<<<dim3(RNN_H / 128, RNN_B / 64), 128, 0, stream>>>(
        xsrc, sx, Wih_b, RNN_D,
        hsrc, sh, Whh_b, RNN_H,
        b_ih, b_hh,
        h_t, sHid, /*tanh=*/1);

    // p_t = h_t W_out^T + b_out                            [4096 x 1024]
    rnn_step_gemm<<<dim3(RNN_D / 128, RNN_B / 64), 128, 0, stream>>>(
        h_t, sHid, Wout_b, RNN_H,
        nullptr, 0, nullptr, 0,
        b_out, nullptr,
        p_t, sRecon, /*tanh=*/0);
  }
}
